// Ernie4_5_VisionAttention_14147622273401
// MI455X (gfx1250) — matrix operand order repaired and checked by reading, not compiled, not run on hardware
//
#include <hip/hip_runtime.h>
#include <hip/hip_bf16.h>
#include <math.h>

typedef __bf16 bf16_t;
typedef __attribute__((ext_vector_type(16))) __bf16 v16bf;
typedef __attribute__((ext_vector_type(8)))  __bf16 v8bf;
typedef __attribute__((ext_vector_type(8)))  float  v8f;

#define SEQ   2048
#define EMB   1280
#define NH    16
#define HD    80
#define DP    96      // head_dim padded to multiple of 32 for WMMA K-loop
#define ROT   40

union BF16x16 { v16bf v; v8bf h[2]; };

static __device__ __forceinline__ v8f wmma_bf16(const v16bf& a, const v16bf& b, const v8f& c) {
  // D = A(16x32 bf16) * B(32x16 bf16) + C(16x16 f32)
  return __builtin_amdgcn_wmma_f32_16x16x32_bf16(false, a, false, b, (short)0, c, false, false);
}

static __device__ __forceinline__ v8bf zero_v8bf() {
  v8bf z;
#pragma unroll
  for (int i = 0; i < 8; ++i) z[i] = (bf16_t)0.0f;
  return z;
}

// CDNA5 async global->LDS copy (16B per enabled lane), tracked by ASYNCcnt.
static __device__ __forceinline__ void async_copy16(unsigned int lds_off, const void* gptr) {
  asm volatile("global_load_async_to_lds_b128 %0, %1, off"
               :: "v"(lds_off), "v"(gptr) : "memory");
}
static __device__ __forceinline__ void wait_async0() {
  asm volatile("s_wait_asynccnt 0x0" ::: "memory");
}

// ---------------------------------------------------------------------------
// fp32 -> bf16 conversion
// ---------------------------------------------------------------------------
__global__ __launch_bounds__(256)
void cvt_f32_bf16(const float* __restrict__ s, bf16_t* __restrict__ d, int n) {
  int i = blockIdx.x * blockDim.x + threadIdx.x;
  if (i < n) d[i] = (bf16_t)s[i];
}

// ---------------------------------------------------------------------------
// C[M,N] = A[M,K](bf16) * W[N,K]^T(bf16) + bias[N]   (fp32 out)
// grid (N/64, M/128), block 128 (4 waves); wave owns a 32x64 strip.
// Double-buffered LDS (one barrier per k-step); A tile staged with
// GLOBAL_LOAD_ASYNC_TO_LDS_B128, B tile transposed through registers.
// ---------------------------------------------------------------------------
__global__ __launch_bounds__(128)
void gemm_bf16_kernel(const bf16_t* __restrict__ A, const bf16_t* __restrict__ W,
                      const float* __restrict__ bias, float* __restrict__ C,
                      int M, int N, int K)
{
  __shared__ __align__(16) bf16_t As[2][128][32];   // [buf][m][k]  8KB each
  __shared__ __align__(16) bf16_t Bt[2][32][64];    // [buf][k][n]  4KB each
  const int tid = threadIdx.x;
  const int wv = tid >> 5, lane = tid & 31;
  const int hf = lane >> 4, ln = lane & 15;
  const int m0 = blockIdx.y * 128, n0 = blockIdx.x * 64;
  const int brow = tid >> 1, bh = tid & 1;

  v8f acc[2][4];
#pragma unroll
  for (int mm = 0; mm < 2; ++mm)
#pragma unroll
    for (int j = 0; j < 4; ++j)
#pragma unroll
      for (int r = 0; r < 8; ++r) acc[mm][j][r] = 0.0f;

  // ---- prologue: async A(0) -> buf0, B(0) -> regs ----
  {
    const bf16_t* ap = &A[(size_t)(m0 + tid) * K];
#pragma unroll
    for (int c = 0; c < 4; ++c)
      async_copy16((unsigned int)(uintptr_t)&As[0][tid][c * 8], ap + c * 8);
  }
  const bf16_t* wp = &W[(size_t)(n0 + brow) * K + bh * 16];
  v8bf w0 = *(const v8bf*)(wp);
  v8bf w1 = *(const v8bf*)(wp + 8);

  int p = 0;
  for (int kk = 0; kk < K; kk += 32, p ^= 1) {
    // commit B(kk) regs -> Bt[p] (transpose scatter)
#pragma unroll
    for (int e = 0; e < 8; ++e) Bt[p][bh * 16 + e][brow]     = w0[e];
#pragma unroll
    for (int e = 0; e < 8; ++e) Bt[p][bh * 16 + 8 + e][brow] = w1[e];
    wait_async0();        // A(kk) landed in As[p]
    __syncthreads();      // publish tile kk; also retires reads of buf p^1

    if (kk + 32 < K) {    // stage tile kk+1 into buf p^1 (overlaps WMMAs below)
      const bf16_t* ap = &A[(size_t)(m0 + tid) * K + kk + 32];
#pragma unroll
      for (int c = 0; c < 4; ++c)
        async_copy16((unsigned int)(uintptr_t)&As[p ^ 1][tid][c * 8], ap + c * 8);
      const bf16_t* wp2 = &W[(size_t)(n0 + brow) * K + kk + 32 + bh * 16];
      w0 = *(const v8bf*)(wp2);
      w1 = *(const v8bf*)(wp2 + 8);
    }

    // A fragment: lane<16 -> M=ln, K {0..7,16..23}; lane>=16 -> K {8..15,24..31}
    BF16x16 af[2];
#pragma unroll
    for (int mm = 0; mm < 2; ++mm) {
      af[mm].h[0] = *(const v8bf*)&As[p][wv * 32 + mm * 16 + ln][hf * 8];
      af[mm].h[1] = *(const v8bf*)&As[p][wv * 32 + mm * 16 + ln][16 + hf * 8];
    }
#pragma unroll
    for (int j = 0; j < 4; ++j) {
      // B fragment: lane -> N=ln, K {0..7,16..23} (lane<16) / {8..15,24..31} (lane>=16)
      BF16x16 bfc;
#pragma unroll
      for (int e = 0; e < 8; ++e) {
        bfc.h[0][e] = Bt[p][hf * 8 + e][16 * j + ln];
        bfc.h[1][e] = Bt[p][16 + hf * 8 + e][16 * j + ln];
      }
#pragma unroll
      for (int mm = 0; mm < 2; ++mm)
        acc[mm][j] = wmma_bf16(af[mm].v, bfc.v, acc[mm][j]);
    }
  }

#pragma unroll
  for (int mm = 0; mm < 2; ++mm)
#pragma unroll
    for (int j = 0; j < 4; ++j) {
      const int col = n0 + 16 * j + ln;
      const float b = bias[col];
#pragma unroll
      for (int r = 0; r < 8; ++r) {
        const int row = m0 + wv * 32 + mm * 16 + hf * 8 + r; // C: VGPR r -> M=r / 8+r
        C[(size_t)row * N + col] = acc[mm][j][r] + b;
      }
    }
}

// ---------------------------------------------------------------------------
// rotary + repack qkv[S,3840](f32) -> q/k/v [head][seq][96](bf16), q pre-scaled
// ---------------------------------------------------------------------------
__global__ __launch_bounds__(96)
void rotary_pack_kernel(const float* __restrict__ qkv, const float* __restrict__ rope,
                        bf16_t* __restrict__ qd, bf16_t* __restrict__ kd,
                        bf16_t* __restrict__ vd)
{
  const int h = blockIdx.x, s = blockIdx.y, d = threadIdx.x;
  const size_t out = ((size_t)h * SEQ + s) * DP + d;
  if (d >= HD) { qd[out] = (bf16_t)0.0f; kd[out] = (bf16_t)0.0f; vd[out] = (bf16_t)0.0f; return; }
  const float scale = 0.11180339887498948f;  // 80^-0.5 folded into q
  const size_t base = (size_t)s * (3 * EMB) + h * HD;
  const float q = qkv[base + d];
  const float k = qkv[base + EMB + d];
  const float v = qkv[base + 2 * EMB + d];
  const int dr = (d < ROT) ? d : d - ROT;
  const float ang = rope[(size_t)s * ROT + dr];
  const float c = __cosf(ang), sn = __sinf(ang);
  float qo, ko;
  if (d < ROT) {
    qo = q * c - qkv[base + d + ROT] * sn;
    ko = k * c - qkv[base + EMB + d + ROT] * sn;
  } else {
    qo = q * c + qkv[base + d - ROT] * sn;
    ko = k * c + qkv[base + EMB + d - ROT] * sn;
  }
  qd[out] = (bf16_t)(qo * scale);
  kd[out] = (bf16_t)ko;
  vd[out] = (bf16_t)v;
}

// ---------------------------------------------------------------------------
// flash attention per (head, segment, 64-query block); 4 waves x 16 rows
// ---------------------------------------------------------------------------
__global__ __launch_bounds__(128)
void attn_kernel(const bf16_t* __restrict__ qg, const bf16_t* __restrict__ kg,
                 const bf16_t* __restrict__ vg, const int* __restrict__ cu,
                 bf16_t* __restrict__ ctx)
{
  const int qblk = blockIdx.x, seg = blockIdx.y, h = blockIdx.z;
  const int s0 = cu[seg], s1 = cu[seg + 1];
  const int seglen = s1 - s0;
  const int q0 = qblk * 64;
  if (q0 >= seglen) return;

  const int tid = threadIdx.x;
  const int wv = tid >> 5, lane = tid & 31;
  const int hf = lane >> 4, ln = lane & 15;

  __shared__ __align__(16) bf16_t Qs[64][DP];      // 12KB
  __shared__ __align__(16) bf16_t Kt[DP][64];      // 12KB  [d][key]
  __shared__ __align__(16) bf16_t Vs[64][DP];      // 12KB  [key][d] (natural B layout)
  __shared__ __align__(16) bf16_t Ps[4][16][64];   //  8KB  per-wave P tile

  // ---- stage Q tile (async copies; zero-fill rows past segment end) ----
  {
    const int row = tid >> 1, h48 = tid & 1;
    const int grow = s0 + q0 + row;
    const bool ok = (q0 + row) < seglen;
    if (ok) {
      const bf16_t* src = &qg[((size_t)h * SEQ + grow) * DP + h48 * 48];
#pragma unroll
      for (int c = 0; c < 6; ++c)
        async_copy16((unsigned int)(uintptr_t)&Qs[row][h48 * 48 + c * 8], src + c * 8);
    } else {
#pragma unroll
      for (int c = 0; c < 6; ++c)
        *(v8bf*)&Qs[row][h48 * 48 + c * 8] = zero_v8bf();
    }
  }
  wait_async0();
  __syncthreads();

  // ---- Q A-fragments, persistent across key tiles ----
  BF16x16 qf[3];
#pragma unroll
  for (int kc = 0; kc < 3; ++kc) {
    qf[kc].h[0] = *(const v8bf*)&Qs[wv * 16 + ln][kc * 32 + hf * 8];
    qf[kc].h[1] = *(const v8bf*)&Qs[wv * 16 + ln][kc * 32 + 16 + hf * 8];
  }

  float mrow[8], lrow[8];
  v8f O[6];
#pragma unroll
  for (int r = 0; r < 8; ++r) { mrow[r] = -INFINITY; lrow[r] = 0.0f; }
#pragma unroll
  for (int t = 0; t < 6; ++t)
#pragma unroll
    for (int r = 0; r < 8; ++r) O[t][r] = 0.0f;

  for (int kt0 = 0; kt0 < seglen; kt0 += 64) {
    __syncthreads();  // previous-iteration Vs readers done before restaging
    // ---- stage K^T (register transpose) and V (async) tiles ----
    {
      const int key = tid >> 1, h48 = tid & 1;
      const int gk = s0 + kt0 + key;
      const bool ok = (kt0 + key) < seglen;
      const bf16_t* ksrc = &kg[((size_t)h * SEQ + gk) * DP + h48 * 48];
      if (ok) {
        const bf16_t* vsrc = &vg[((size_t)h * SEQ + gk) * DP + h48 * 48];
#pragma unroll
        for (int c = 0; c < 6; ++c)
          async_copy16((unsigned int)(uintptr_t)&Vs[key][h48 * 48 + c * 8], vsrc + c * 8);
      } else {
#pragma unroll
        for (int c = 0; c < 6; ++c)
          *(v8bf*)&Vs[key][h48 * 48 + c * 8] = zero_v8bf();
      }
#pragma unroll
      for (int c = 0; c < 6; ++c) {
        v8bf tk = ok ? *(const v8bf*)(ksrc + c * 8) : zero_v8bf();
#pragma unroll
        for (int e = 0; e < 8; ++e) Kt[h48 * 48 + c * 8 + e][key] = tk[e];
      }
    }
    wait_async0();
    __syncthreads();

    // ---- S = Q K^T (16x64 per wave, scale pre-folded into Q) ----
    v8f S[4];
#pragma unroll
    for (int j = 0; j < 4; ++j)
#pragma unroll
      for (int r = 0; r < 8; ++r) S[j][r] = 0.0f;
#pragma unroll
    for (int kc = 0; kc < 3; ++kc) {
#pragma unroll
      for (int j = 0; j < 4; ++j) {
        BF16x16 bfc;
#pragma unroll
        for (int e = 0; e < 8; ++e) {
          bfc.h[0][e] = Kt[kc * 32 + hf * 8 + e][16 * j + ln];
          bfc.h[1][e] = Kt[kc * 32 + 16 + hf * 8 + e][16 * j + ln];
        }
        S[j] = wmma_bf16(qf[kc].v, bfc.v, S[j]);
      }
    }

    // ---- mask columns past segment end ----
#pragma unroll
    for (int j = 0; j < 4; ++j) {
      if (kt0 + 16 * j + ln >= seglen) {
#pragma unroll
        for (int r = 0; r < 8; ++r) S[j][r] = -INFINITY;
      }
    }

    // ---- online softmax; P tile -> LDS (C-layout write) ----
#pragma unroll
    for (int r = 0; r < 8; ++r) {
      float mx = fmaxf(fmaxf(S[0][r], S[1][r]), fmaxf(S[2][r], S[3][r]));
#pragma unroll
      for (int off = 1; off < 16; off <<= 1) mx = fmaxf(mx, __shfl_xor(mx, off, 16));
      const float mnew = fmaxf(mrow[r], mx);
      const float corr = __expf(mrow[r] - mnew);
      mrow[r] = mnew;
      float rs = 0.0f;
#pragma unroll
      for (int j = 0; j < 4; ++j) {
        const float p = __expf(S[j][r] - mnew);
        rs += p;
        Ps[wv][hf * 8 + r][16 * j + ln] = (bf16_t)p;
      }
#pragma unroll
      for (int off = 1; off < 16; off <<= 1) rs += __shfl_xor(rs, off, 16);
      lrow[r] = lrow[r] * corr + rs;
#pragma unroll
      for (int t = 0; t < 6; ++t) O[t][r] *= corr;
    }
    __syncthreads();  // P visible for A-layout re-read

    // ---- O += P V ----
#pragma unroll
    for (int kc = 0; kc < 2; ++kc) {
      BF16x16 pf;
      pf.h[0] = *(const v8bf*)&Ps[wv][ln][kc * 32 + hf * 8];
      pf.h[1] = *(const v8bf*)&Ps[wv][ln][kc * 32 + 16 + hf * 8];
#pragma unroll
      for (int t = 0; t < 6; ++t) {
        BF16x16 vfc;
#pragma unroll
        for (int e = 0; e < 8; ++e) {
          vfc.h[0][e] = Vs[kc * 32 + hf * 8 + e][16 * t + ln];
          vfc.h[1][e] = Vs[kc * 32 + 16 + hf * 8 + e][16 * t + ln];
        }
        O[t] = wmma_bf16(pf.v, vfc.v, O[t]);
      }
    }
  }

  // ---- normalize and store ctx (drop the 16 padded dims) ----
#pragma unroll
  for (int r = 0; r < 8; ++r) {
    const int row = s0 + q0 + wv * 16 + hf * 8 + r;
    if (row < s1) {
      const float inv = 1.0f / lrow[r];
#pragma unroll
      for (int t = 0; t < 5; ++t) {
        ctx[(size_t)row * EMB + h * HD + 16 * t + ln] = (bf16_t)(O[t][r] * inv);
      }
    }
  }
}

// ---------------------------------------------------------------------------
extern "C" void kernel_launch(void* const* d_in, const int* in_sizes, int n_in,
                              void* d_out, int out_size, void* d_ws, size_t ws_size,
                              hipStream_t stream) {
  const float* x      = (const float*)d_in[0];
  const float* rope   = (const float*)d_in[1];
  const int*   cu     = (const int*)d_in[2];
  const float* qkv_w  = (const float*)d_in[4];
  const float* qkv_b  = (const float*)d_in[5];
  const float* proj_w = (const float*)d_in[6];
  const float* proj_b = (const float*)d_in[7];

  char* ws = (char*)d_ws;
  size_t off = 0;
  auto take = [&](size_t bytes) -> void* {
    void* p = ws + off;
    off = (off + bytes + 255) & ~(size_t)255;
    return p;
  };
  bf16_t* xbf     = (bf16_t*)take((size_t)SEQ * EMB * 2);
  bf16_t* qkvwbf  = (bf16_t*)take((size_t)3 * EMB * EMB * 2);
  bf16_t* projwbf = (bf16_t*)take((size_t)EMB * EMB * 2);
  float*  qkvf    = (float*) take((size_t)SEQ * 3 * EMB * 4);
  bf16_t* qbf     = (bf16_t*)take((size_t)NH * SEQ * DP * 2);
  bf16_t* kbf     = (bf16_t*)take((size_t)NH * SEQ * DP * 2);
  bf16_t* vbf     = (bf16_t*)take((size_t)NH * SEQ * DP * 2);
  bf16_t* ctxbf   = (bf16_t*)take((size_t)SEQ * EMB * 2);

  // 1. convert operands to bf16
  {
    int n0 = SEQ * EMB, n1 = 3 * EMB * EMB, n2 = EMB * EMB;
    cvt_f32_bf16<<<(n0 + 255) / 256, 256, 0, stream>>>(x, xbf, n0);
    cvt_f32_bf16<<<(n1 + 255) / 256, 256, 0, stream>>>(qkv_w, qkvwbf, n1);
    cvt_f32_bf16<<<(n2 + 255) / 256, 256, 0, stream>>>(proj_w, projwbf, n2);
  }
  // 2. QKV projection (2048 x 3840, K=1280)
  gemm_bf16_kernel<<<dim3((3 * EMB) / 64, SEQ / 128), 128, 0, stream>>>(
      xbf, qkvwbf, qkv_b, qkvf, SEQ, 3 * EMB, EMB);
  // 3. rotary + repack to [head][seq][96] bf16
  rotary_pack_kernel<<<dim3(NH, SEQ), 96, 0, stream>>>(qkvf, rope, qbf, kbf, vbf);
  // 4. block-diagonal flash attention (8 qblocks x 4 segments x 16 heads)
  attn_kernel<<<dim3(8, 4, NH), 128, 0, stream>>>(qbf, kbf, vbf, cu, ctxbf);
  // 5. output projection into d_out (fp32)
  gemm_bf16_kernel<<<dim3(EMB / 64, SEQ / 128), 128, 0, stream>>>(
      ctxbf, projwbf, proj_b, (float*)d_out, SEQ, EMB, EMB);
}
